// Transformer_31568009625958
// MI455X (gfx1250) — matrix operand order repaired and checked by reading, not compiled, not run on hardware
//
#include <hip/hip_runtime.h>
#include <hip/hip_bf16.h>
#include <math.h>

// ---- model dims (fixed by reference) ----
#define BATCH 2
#define SEQ   2048
#define DIM   512
#define HEADS 8
#define DH    64
#define MLPD  2048
#define ROWS  (BATCH * SEQ)           // 4096
#define QKVW  (3 * DIM)               // 1536
#define SCALE 0.125f                  // DH^-0.5

typedef __attribute__((ext_vector_type(16))) __bf16 v16bf;
typedef __attribute__((ext_vector_type(8)))  float  v8f;

union FragBF {
  v16bf v;
  uint4 q[2];
  __bf16 h[16];
};

static __device__ __forceinline__ v8f wmma_bf16(v16bf a, v16bf b, v8f c) {
  return __builtin_amdgcn_wmma_f32_16x16x32_bf16(false, a, false, b, (short)0, c,
                                                 false, false);
}

// Async global->LDS 16B copy (gfx1250, tracked by ASYNCcnt)
static __device__ __forceinline__ void async_copy_b128(void* lds, const void* g) {
  asm volatile("global_load_async_to_lds_b128 %0, %1, off"
               :
               : "v"((unsigned)(size_t)lds), "v"((unsigned long long)(size_t)g)
               : "memory");
}
static __device__ __forceinline__ void async_wait0() {
  asm volatile("s_wait_asynccnt 0x0" ::: "memory");
}

// ---- 16-lane (DPP row) reductions: pure VALU, no LDS traffic ----
#define DPP_QX1 0xB1   // quad_perm [1,0,3,2]
#define DPP_QX2 0x4E   // quad_perm [2,3,0,1]
#define DPP_RHM 0x141  // row_half_mirror
#define DPP_RM  0x140  // row_mirror

#define DPPF(x, ctrl)                                                      \
  __builtin_bit_cast(float, __builtin_amdgcn_mov_dpp(                      \
                                __builtin_bit_cast(int, (x)), (ctrl), 0xF, \
                                0xF, true))

static __device__ __forceinline__ float red16_max(float v) {
  v = fmaxf(v, DPPF(v, DPP_QX1));
  v = fmaxf(v, DPPF(v, DPP_QX2));
  v = fmaxf(v, DPPF(v, DPP_RHM));
  v = fmaxf(v, DPPF(v, DPP_RM));
  return v;
}
static __device__ __forceinline__ float red16_add(float v) {
  v += DPPF(v, DPP_QX1);
  v += DPPF(v, DPP_QX2);
  v += DPPF(v, DPP_RHM);
  v += DPPF(v, DPP_RM);
  return v;
}

// ---------------------------------------------------------------------------
__global__ __launch_bounds__(256) void copy_kernel(const float* __restrict__ s,
                                                   float* __restrict__ d, int n) {
  int i = blockIdx.x * 256 + threadIdx.x;
  if (i < n) d[i] = s[i];
}

// ---------------------------------------------------------------------------
// LayerNorm: one wave per row of 512, fp32 in -> bf16 out
// ---------------------------------------------------------------------------
__global__ __launch_bounds__(256) void ln_kernel(const float* __restrict__ X,
                                                 const float* __restrict__ G,
                                                 const float* __restrict__ Bb,
                                                 __bf16* __restrict__ O) {
  const int wid = threadIdx.x >> 5, lane = threadIdx.x & 31;
  const int row = blockIdx.x * 8 + wid;
  const float* x = X + (size_t)row * DIM + lane * 16;
  float vals[16];
  const float4* p = reinterpret_cast<const float4*>(x);
#pragma unroll
  for (int i = 0; i < 4; ++i) {
    float4 f = p[i];
    vals[4 * i + 0] = f.x; vals[4 * i + 1] = f.y;
    vals[4 * i + 2] = f.z; vals[4 * i + 3] = f.w;
  }
  float s = 0.f;
#pragma unroll
  for (int i = 0; i < 16; ++i) s += vals[i];
#pragma unroll
  for (int m = 16; m >= 1; m >>= 1) s += __shfl_xor(s, m, 32);
  const float mean = s * (1.0f / DIM);
  float ss = 0.f;
#pragma unroll
  for (int i = 0; i < 16; ++i) { float d = vals[i] - mean; ss += d * d; }
#pragma unroll
  for (int m = 16; m >= 1; m >>= 1) ss += __shfl_xor(ss, m, 32);
  const float rstd = rsqrtf(ss * (1.0f / DIM) + 1e-5f);
#pragma unroll
  for (int i = 0; i < 16; ++i) {
    int col = lane * 16 + i;
    O[(size_t)row * DIM + col] = (__bf16)((vals[i] - mean) * rstd * G[col] + Bb[col]);
  }
}

// ---------------------------------------------------------------------------
// GEMM: Out[M,N] = A_bf16[M,K] @ W_f32[K,N]  (+epilogue)
// MODE 0: plain -> bf16 out          (QKV)
// MODE 1: +bias, +residual -> f32 h  (attn out-proj)
// MODE 2: +bias, gelu -> bf16 out    (MLP fc1)
// MODE 3: +bias, +residual -> f32 h  (MLP fc2)
// block tile 128x128, 8 waves each 32x64 (2x4 wmma tiles), K staged by 32.
// A staged [M][K] via async global->LDS b128; W staged transposed [N][K]
// (fp32->bf16 on the fly) so all fragment reads are ds_load_b128.
// ---------------------------------------------------------------------------
template <int MODE>
__global__ __launch_bounds__(256) void gemm_bf16(const __bf16* __restrict__ A,
                                                 const float* __restrict__ W,
                                                 const float* __restrict__ bias,
                                                 float* __restrict__ H,
                                                 __bf16* __restrict__ OutB,
                                                 int K, int N) {
  __shared__ __bf16 As[128][40];    // row stride 80B (16B aligned)
  __shared__ __bf16 WsT[128][40];   // transposed: [col][k]
  const int t = threadIdx.x, lane = t & 31, wid = t >> 5;
  const int bm = blockIdx.x * 128, bn = blockIdx.y * 128;
  const int wm = (wid >> 1) * 32, wn = (wid & 1) * 64;
  const int m16 = lane & 15, kh = lane >> 4;
  v8f c[2][4] = {};

  const int ai = t >> 1, aj = (t & 1) * 16;       // A staging coords
  const int wcol = t & 127, wks = (t >> 7) * 16;  // W staging coords

  for (int k0 = 0; k0 < K; k0 += 32) {
    {  // stage A tile [128][32]: 2 async b128 copies per thread
      const __bf16* src = A + (size_t)(bm + ai) * K + k0 + aj;
      async_copy_b128(&As[ai][aj], src);
      async_copy_b128(&As[ai][aj + 8], src + 8);
    }
    {  // stage W tile transposed: coalesced fp32 loads, b128 LDS stores
#pragma unroll
      for (int g = 0; g < 2; ++g) {
        FragBF u;
#pragma unroll
        for (int kk = 0; kk < 8; ++kk)
          u.h[kk] = (__bf16)W[(size_t)(k0 + wks + g * 8 + kk) * N + bn + wcol];
        *reinterpret_cast<uint4*>(&WsT[wcol][wks + g * 8]) = u.q[0];
      }
    }
    if (k0 + 32 < K)  // prefetch next K-step W lines
      __builtin_prefetch(W + (size_t)(k0 + 32 + wks) * N + bn + wcol, 0, 1);
    async_wait0();
    __syncthreads();
    FragBF a0, a1;
    a0.q[0] = *reinterpret_cast<const uint4*>(&As[wm + m16][kh * 8]);
    a0.q[1] = *reinterpret_cast<const uint4*>(&As[wm + m16][16 + kh * 8]);
    a1.q[0] = *reinterpret_cast<const uint4*>(&As[wm + 16 + m16][kh * 8]);
    a1.q[1] = *reinterpret_cast<const uint4*>(&As[wm + 16 + m16][16 + kh * 8]);
#pragma unroll
    for (int ct = 0; ct < 4; ++ct) {
      FragBF b;
      b.q[0] = *reinterpret_cast<const uint4*>(&WsT[wn + ct * 16 + m16][kh * 8]);
      b.q[1] =
          *reinterpret_cast<const uint4*>(&WsT[wn + ct * 16 + m16][16 + kh * 8]);
      c[0][ct] = wmma_bf16(a0.v, b.v, c[0][ct]);
      c[1][ct] = wmma_bf16(a1.v, b.v, c[1][ct]);
    }
    __syncthreads();
  }

#pragma unroll
  for (int rt = 0; rt < 2; ++rt) {
#pragma unroll
    for (int ct = 0; ct < 4; ++ct) {
#pragma unroll
      for (int r = 0; r < 8; ++r) {
        const int row = bm + wm + rt * 16 + r + kh * 8;
        const int col = bn + wn + ct * 16 + m16;
        float v = c[rt][ct][r];
        if (MODE != 0) v += bias[col];
        if (MODE == 2) v = 0.5f * v * (1.0f + erff(v * 0.70710678f));
        if (MODE == 0 || MODE == 2) {
          OutB[(size_t)row * N + col] = (__bf16)v;
        } else {
          const size_t idx = (size_t)row * N + col;
          H[idx] = H[idx] + v;
        }
      }
    }
  }
}

// ---------------------------------------------------------------------------
// Flash-style masked attention with renormalization:
//   z = Acc / (F + 1e-10*E),  e=exp(s-M), E=sum e, F=sum e*m, Acc=sum e*m*v
// 4 waves/block; wave owns 16 queries; key chunks of 32 staged in LDS.
// K staged [key][d] via async b128 (contraction-contiguous for Q·K^T);
// V staged transposed [d][key] with packed b32 stores; P per-wave [q][key].
// ---------------------------------------------------------------------------
__global__ __launch_bounds__(128) void attn_kernel(const __bf16* __restrict__ qkv,
                                                   const int* __restrict__ mask,
                                                   __bf16* __restrict__ Z) {
  __shared__ __bf16 Ks[32][72];
  __shared__ __bf16 VsT[64][40];
  __shared__ __bf16 Ps[4][16][40];
  const int t = threadIdx.x, lane = t & 31, wid = t >> 5;
  const int blk = blockIdx.x;
  const int qb = blk & 31, hh = (blk >> 5) & 7, b = blk >> 8;
  const int qbase = qb * 64 + wid * 16;
  const int m16 = lane & 15, kh = lane >> 4;

  // Q fragments: 4 x 16B global loads (A-layout is contiguous per lane)
  FragBF a0, a1;
  {
    const __bf16* qp = qkv + (size_t)(b * SEQ + qbase + m16) * QKVW + hh * DH;
    a0.q[0] = *reinterpret_cast<const uint4*>(qp + kh * 8);
    a0.q[1] = *reinterpret_cast<const uint4*>(qp + 16 + kh * 8);
    a1.q[0] = *reinterpret_cast<const uint4*>(qp + 32 + kh * 8);
    a1.q[1] = *reinterpret_cast<const uint4*>(qp + 48 + kh * 8);
  }
  float Ms[8], Es[8], Fs[8];
  v8f acc[4] = {};
#pragma unroll
  for (int r = 0; r < 8; ++r) { Ms[r] = -1e30f; Es[r] = 0.f; Fs[r] = 0.f; }

  const int kkey = t >> 2, kd0 = (t & 3) * 16;    // K staging coords
  const int vkey = (t >> 3) * 2, vd0 = (t & 7) * 8;  // V staging coords

  for (int kc = 0; kc < SEQ / 32; ++kc) {
    const int kbase = kc * 32;
    {  // K: async global->LDS (2 x b128 per thread)
      const __bf16* src =
          qkv + (size_t)(b * SEQ + kbase + kkey) * QKVW + DIM + hh * DH + kd0;
      async_copy_b128(&Ks[kkey][kd0], src);
      async_copy_b128(&Ks[kkey][kd0 + 8], src + 8);
    }
    {  // V: transpose-stage a key-pair x 8 d; packed 4B LDS stores
      const size_t r0 = (size_t)(b * SEQ + kbase + vkey) * QKVW + 2 * DIM +
                        hh * DH + vd0;
      FragBF ua, ub;
      ua.q[0] = *reinterpret_cast<const uint4*>(qkv + r0);
      ub.q[0] = *reinterpret_cast<const uint4*>(qkv + r0 + QKVW);
#pragma unroll
      for (int i = 0; i < 8; ++i) {
        union { __bf16 h2[2]; unsigned u; } w;
        w.h2[0] = ua.h[i];
        w.h2[1] = ub.h[i];
        *reinterpret_cast<unsigned*>(&VsT[vd0 + i][vkey]) = w.u;
      }
    }
    async_wait0();
    __syncthreads();

    // S = Q @ K^T for two 16-key subtiles (contraction over dh=64)
    v8f c0 = {}, c1 = {};
#pragma unroll
    for (int sub = 0; sub < 2; ++sub) {
      FragBF bk0, bk1;
      bk0.q[0] = *reinterpret_cast<const uint4*>(&Ks[sub * 16 + m16][kh * 8]);
      bk0.q[1] = *reinterpret_cast<const uint4*>(&Ks[sub * 16 + m16][16 + kh * 8]);
      bk1.q[0] = *reinterpret_cast<const uint4*>(&Ks[sub * 16 + m16][32 + kh * 8]);
      bk1.q[1] =
          *reinterpret_cast<const uint4*>(&Ks[sub * 16 + m16][48 + kh * 8]);
      v8f c = {};
      c = wmma_bf16(a0.v, bk0.v, c);
      c = wmma_bf16(a1.v, bk1.v, c);
      if (sub == 0) c0 = c; else c1 = c;
    }

    // online softmax stats per row (row = r + 8*kh within tile)
#pragma unroll
    for (int r = 0; r < 8; ++r) {
      const float s0 = c0[r] * SCALE;
      const float s1 = c1[r] * SCALE;
      const float mx = red16_max(fmaxf(s0, s1));
      const float newM = fmaxf(Ms[r], mx);
      const float scl = expf(Ms[r] - newM);
      const float p0 = expf(s0 - newM), p1 = expf(s1 - newM);
      const int qrow = qbase + r + kh * 8;
      const size_t mb = ((size_t)b * SEQ + qrow) * SEQ + kbase;
      const float pm0 = p0 * (float)mask[mb + m16];
      const float pm1 = p1 * (float)mask[mb + 16 + m16];
      const float rs = red16_add(p0 + p1);
      const float rsm = red16_add(pm0 + pm1);
      Es[r] = Es[r] * scl + rs;
      Fs[r] = Fs[r] * scl + rsm;
      Ms[r] = newM;
#pragma unroll
      for (int j = 0; j < 4; ++j) acc[j][r] *= scl;
      Ps[wid][r + kh * 8][m16] = (__bf16)pm0;
      Ps[wid][r + kh * 8][16 + m16] = (__bf16)pm1;
    }

    // Acc += P @ V (contraction over 32 keys)
    FragBF ap;
    ap.q[0] = *reinterpret_cast<const uint4*>(&Ps[wid][m16][kh * 8]);
    ap.q[1] = *reinterpret_cast<const uint4*>(&Ps[wid][m16][16 + kh * 8]);
#pragma unroll
    for (int j = 0; j < 4; ++j) {
      FragBF bv;
      bv.q[0] = *reinterpret_cast<const uint4*>(&VsT[j * 16 + m16][kh * 8]);
      bv.q[1] = *reinterpret_cast<const uint4*>(&VsT[j * 16 + m16][16 + kh * 8]);
      acc[j] = wmma_bf16(ap.v, bv.v, acc[j]);
    }
    __syncthreads();
  }

  // z = Acc / (F + 1e-10*E), head-major layout [B*L, HEADS*DH]
#pragma unroll
  for (int r = 0; r < 8; ++r) {
    const int qrow = qbase + r + kh * 8;
    const float inv = 1.0f / (Fs[r] + 1e-10f * Es[r]);
#pragma unroll
    for (int j = 0; j < 4; ++j) {
      Z[(size_t)(b * SEQ + qrow) * DIM + hh * DH + j * 16 + m16] =
          (__bf16)(acc[j][r] * inv);
    }
  }
}

// ---------------------------------------------------------------------------
extern "C" void kernel_launch(void* const* d_in, const int* in_sizes, int n_in,
                              void* d_out, int out_size, void* d_ws,
                              size_t ws_size, hipStream_t stream) {
  const float* x     = (const float*)d_in[0];
  const int*   mask  = (const int*)d_in[1];
  const float* ln1_g = (const float*)d_in[2];
  const float* ln1_b = (const float*)d_in[3];
  const float* w_qkv = (const float*)d_in[4];
  const float* w_out = (const float*)d_in[5];
  const float* b_out = (const float*)d_in[6];
  const float* ln2_g = (const float*)d_in[7];
  const float* ln2_b = (const float*)d_in[8];
  const float* w1    = (const float*)d_in[9];
  const float* b1    = (const float*)d_in[10];
  const float* w2    = (const float*)d_in[11];
  const float* b2    = (const float*)d_in[12];

  float* h = (float*)d_out;  // residual stream lives in d_out

  char* ws = (char*)d_ws;
  __bf16* lnbuf  = (__bf16*)(ws);                                 // 4 MB
  __bf16* qkvbuf = (__bf16*)(ws + (size_t)ROWS * DIM * 2);        // 12 MB
  __bf16* zbuf   = (__bf16*)(ws + (size_t)ROWS * (DIM + QKVW) * 2);
  __bf16* midbuf = (__bf16*)(ws + (size_t)ROWS * (2 * DIM + QKVW) * 2);

  const int nElem = ROWS * DIM;
  copy_kernel<<<(nElem + 255) / 256, 256, 0, stream>>>(x, h, nElem);

  for (int d = 0; d < 2; ++d) {
    ln_kernel<<<ROWS / 8, 256, 0, stream>>>(h, ln1_g + d * DIM, ln1_b + d * DIM,
                                            lnbuf);
    gemm_bf16<0><<<dim3(ROWS / 128, QKVW / 128), 256, 0, stream>>>(
        lnbuf, w_qkv + (size_t)d * DIM * QKVW, nullptr, nullptr, qkvbuf, DIM,
        QKVW);
    attn_kernel<<<BATCH * HEADS * (SEQ / 64), 128, 0, stream>>>(qkvbuf, mask,
                                                                zbuf);
    gemm_bf16<1><<<dim3(ROWS / 128, DIM / 128), 256, 0, stream>>>(
        zbuf, w_out + (size_t)d * DIM * DIM, b_out + d * DIM, h, nullptr, DIM,
        DIM);
    ln_kernel<<<ROWS / 8, 256, 0, stream>>>(h, ln2_g + d * DIM, ln2_b + d * DIM,
                                            lnbuf);
    gemm_bf16<2><<<dim3(ROWS / 128, MLPD / 128), 256, 0, stream>>>(
        lnbuf, w1 + (size_t)d * DIM * MLPD, b1 + d * MLPD, nullptr, midbuf, DIM,
        MLPD);
    gemm_bf16<3><<<dim3(ROWS / 128, DIM / 128), 256, 0, stream>>>(
        midbuf, w2 + (size_t)d * MLPD * DIM, b2 + d * DIM, h, nullptr, MLPD,
        DIM);
  }
}
